// Forward_14353780703441
// MI455X (gfx1250) — matrix operand order repaired and checked by reading, not compiled, not run on hardware
//
#include <hip/hip_runtime.h>

#define Bdim 512
#define Tdim 512
#define Kdim 128

typedef __attribute__((ext_vector_type(16))) _Float16 v16h;
typedef __attribute__((ext_vector_type(8)))  _Float16 v8h;
typedef __attribute__((ext_vector_type(4)))  _Float16 v4h;
typedef __attribute__((ext_vector_type(8)))  float    v8f;
typedef __attribute__((ext_vector_type(4)))  float    v4f;

// ---- DPP16 cross-lane reductions over each 16-lane row (pure VALU) --------
template <int CTRL>
__device__ __forceinline__ int dpp_i(int x) {
    return __builtin_amdgcn_update_dpp(x, x, CTRL, 0xF, 0xF, true);
}
template <int CTRL>
__device__ __forceinline__ float dpp_f(float v) {
    return __int_as_float(dpp_i<CTRL>(__float_as_int(v)));
}
// max over 16-lane row for POSITIVE floats: integer-domain compare (no
// canonicalize ops; IEEE order == integer order for non-negative floats).
__device__ __forceinline__ float row16_max_pos(float v) {
    int x = __float_as_int(v);
    x = max(x, dpp_i<0xB1>(x));    // quad_perm [1,0,3,2]  (xor 1)
    x = max(x, dpp_i<0x4E>(x));    // quad_perm [2,3,0,1]  (xor 2)
    x = max(x, dpp_i<0x124>(x));   // row_ror:4
    x = max(x, dpp_i<0x128>(x));   // row_ror:8
    return __int_as_float(x);
}
__device__ __forceinline__ float row16_sum(float v) {
    v += dpp_f<0xB1>(v);
    v += dpp_f<0x4E>(v);
    v += dpp_f<0x124>(v);
    v += dpp_f<0x128>(v);
    return v;
}

// ---------------------------------------------------------------------------
// Kernel 1: gold-path score. One WAVE per batch element; lanes stride over t
// so 16 gathers per lane are in flight; DPP + one 16-swap shuffle to reduce.
// ---------------------------------------------------------------------------
__global__ __launch_bounds__(256) void crf_score_kernel(
    const float* __restrict__ pot, const int* __restrict__ tags,
    const float* __restrict__ trans, float* __restrict__ scores)
{
    const int wave = threadIdx.x >> 5;
    const int lane = threadIdx.x & 31;
    const int b    = blockIdx.x * 8 + wave;
    const float* pb = pot + (size_t)b * Tdim * Kdim;
    const int*   tb = tags + (size_t)b * Tdim;

    float s = 0.f;
    for (int t = lane; t < Tdim; t += 32) {
        int cur = tb[t];
        s += pb[(size_t)t * Kdim + cur];
        if (t > 0) s += trans[tb[t - 1] * Kdim + cur];
    }
    s = row16_sum(s);
    s += __shfl_xor(s, 16, 32);          // combine the two 16-lane halves
    if (lane == 0) scores[b] = s;
}

// ---------------------------------------------------------------------------
// Kernel 2: forward algorithm, fully in exp-space.
//   q_t = exp(alpha_t - m_{t-1});  P = q/qmax;  C = P @ exp(trans) via WMMA;
//   q_{t+1} = C * exp(pot);        L += log(qmax)  (one log per row per step)
//   logZ = L + log(sum_j q_final)
// One WG = 16 batch rows, 8 waves; wave w owns columns [16w,16w+16).
// Pot pipeline: rw = raw pot(t+1) loaded at iter t-1; exp'd at iter t;
// loads for t+2 issued at iter t. Loop unrolled x2 to kill rotation movs.
// ---------------------------------------------------------------------------
__global__ __launch_bounds__(256) void crf_forward_kernel(
    const float* __restrict__ pot, const float* __restrict__ trans,
    const float* __restrict__ scores, float* __restrict__ out)
{
    __shared__ __align__(16) float    Qf[16][132];   // q_t (f32)
    __shared__ __align__(16) _Float16 Ph[16][136];   // P = q/qmax (f16, A-matrix)

    const int tid   = threadIdx.x;
    const int lane  = tid & 31;
    const int wave  = tid >> 5;          // 0..7 : column tile of K
    const int hi    = (lane >> 4) & 1;   // half-wave select
    const int l15   = lane & 15;
    const int row   = tid >> 4;          // 0..15 : row for max/P phases
    const int c0    = tid & 15;
    const int bBase = blockIdx.x * 16;
    const int colW  = 16 * wave + l15;   // this lane's N column (WMMA tile)
    const int cA    = 4 * c0;            // P-phase chunk: cols cA..cA+3, 64+cA..
    const size_t rowStride = (size_t)Tdim * Kdim;
    const int M0   = 8 * hi;             // first C-row this half-wave owns

    const float* potBase = pot + colW;

    // ---- constant B-fragments: E = exp(trans) tile, resident in VGPRs ----
    v16h e[4];
    #pragma unroll
    for (int kk = 0; kk < 4; ++kk) {
        #pragma unroll
        for (int h = 0; h < 16; ++h) {
            int k = 32 * kk + (h < 8 ? 8 * hi + h : 16 + 8 * hi + (h - 8));
            e[kk][h] = (_Float16)__expf(trans[k * Kdim + colW]);
        }
    }

    // ---- init: q_0 = exp(pot[:, 0, :]) ----
    {
        const float* p0 = pot + (size_t)(bBase + row) * rowStride;
        #pragma unroll
        for (int i = 0; i < 4; ++i) {
            Qf[row][cA + i]      = __expf(p0[cA + i]);
            Qf[row][64 + cA + i] = __expf(p0[64 + cA + i]);
        }
    }

    // ---- pipeline prologue: pr = epot(1), rw = raw pot(2) ----
    float pr[8], rw[8];
    #pragma unroll
    for (int r = 0; r < 8; ++r) {
        size_t base = (size_t)(bBase + r + M0) * rowStride;
        pr[r] = __expf(potBase[base + (size_t)Kdim]);
        rw[r] = potBase[base + 2 * (size_t)Kdim];
    }

    float Lrow = 0.f, Lc = 0.f;          // Kahan accumulator: sum_t log(qmax_t)

    #pragma unroll 2
    for (int t = 1; t < Tdim; ++t) {
        __syncthreads();                 // q_{t-1} visible

        // ---- loop-top: qmax (DPP reduction), P = q/qmax -> f16, L += log ----
        const float* qp = &Qf[row][0];
        v4f q0 = *(const v4f*)(qp + cA);
        v4f q1 = *(const v4f*)(qp + 64 + cA);
        float qm = fmaxf(fmaxf(fmaxf(q0[0], q0[1]), fmaxf(q0[2], q0[3])),
                         fmaxf(fmaxf(q1[0], q1[1]), fmaxf(q1[2], q1[3])));
        qm = row16_max_pos(qm);
        float inv = __builtin_amdgcn_rcpf(qm);
        v4h p0v, p1v;
        #pragma unroll
        for (int i = 0; i < 4; ++i) {
            p0v[i] = (_Float16)(q0[i] * inv);
            p1v[i] = (_Float16)(q1[i] * inv);
        }
        *(v4h*)&Ph[row][cA]      = p0v;  // ds_store_b64, packed cvt
        *(v4h*)&Ph[row][64 + cA] = p1v;

        // ---- pot pipeline (overlaps with P phase / barrier / WMMA) ----
        int t2 = t + 2 <= Tdim - 1 ? t + 2 : Tdim - 1;
        int t4 = t + 4 <= Tdim - 1 ? t + 4 : Tdim - 1;
        float rwN[8], pn[8];
        #pragma unroll
        for (int r = 0; r < 8; ++r) {
            size_t base = (size_t)(bBase + r + M0) * rowStride;
            rwN[r] = potBase[base + (size_t)t2 * Kdim];
            __builtin_prefetch(&potBase[base + (size_t)t4 * Kdim], 0, 3);
        }
        #pragma unroll
        for (int r = 0; r < 8; ++r) pn[r] = __expf(rw[r]);   // epot(t+1)

        // one log per row per step, Kahan; all lanes compute (uniform per row,
        // avoids exec save/restore) -- only c0==0's copy is read at the end.
        {
            float y = __logf(qm) - Lc;
            float s2 = Lrow + y;
            Lc = (s2 - Lrow) - y;
            Lrow = s2;
        }

        __syncthreads();                 // Ph visible

        // ---- all 8 A-fragment ds_load_b128 first, fenced, then 4 WMMAs ----
        v16h a[4];
        #pragma unroll
        for (int kk = 0; kk < 4; ++kk) {
            const _Float16* prow = &Ph[l15][32 * kk + 8 * hi];
            v8h lo8 = *(const v8h*)prow;
            v8h hi8 = *(const v8h*)(prow + 16);
            #pragma unroll
            for (int h = 0; h < 8; ++h) { a[kk][h] = lo8[h]; a[kk][h + 8] = hi8[h]; }
        }
        __builtin_amdgcn_sched_barrier(0);   // keep loads grouped: one dscnt wait
        v8f c = {};
        #pragma unroll
        for (int kk = 0; kk < 4; ++kk)
            c = __builtin_amdgcn_wmma_f32_16x16x32_f16(
                    false, a[kk], false, e[kk], (short)0, c, false, false);

        // ---- phase 4: q_t = C * exp(pot_t); rotate the pipeline ----
        #pragma unroll
        for (int r = 0; r < 8; ++r) {
            Qf[r + M0][colW] = c[r] * pr[r];   // C layout: lanes>=16 -> M=r+8
            pr[r] = pn[r];
            rw[r] = rwN[r];
        }
    }

    __syncthreads();
    // ---- final: logZ = Lrow + log(sum_j q_{T-1,j}) ----
    {
        const float* qp = &Qf[row][0];
        v4f q0 = *(const v4f*)(qp + cA);
        v4f q1 = *(const v4f*)(qp + 64 + cA);
        float s = ((q0[0] + q0[1]) + (q0[2] + q0[3]))
                + ((q1[0] + q1[1]) + (q1[2] + q1[3]));
        s = row16_sum(s);
        if (c0 == 0)
            out[bBase + row] = scores[bBase + row] - (Lrow + __logf(s));
    }
}

// ---------------------------------------------------------------------------
extern "C" void kernel_launch(void* const* d_in, const int* in_sizes, int n_in,
                              void* d_out, int out_size, void* d_ws, size_t ws_size,
                              hipStream_t stream) {
    const float* pot   = (const float*)d_in[0];   // output:            [B,T,K] f32
    const int*   tags  = (const int*)  d_in[1];   // label_input:       [B,T]   i32
    const float* trans = (const float*)d_in[2];   // transition_params: [K,K]   f32
    float* outp   = (float*)d_out;                // [B] f32
    float* scores = (float*)d_ws;                 // [B] f32 scratch

    crf_score_kernel<<<Bdim / 8, 256, 0, stream>>>(pot, tags, trans, scores);
    crf_forward_kernel<<<Bdim / 16, 256, 0, stream>>>(pot, trans, scores, outp);
}
